// Generator_8_37752762532141
// MI455X (gfx1250) — matrix operand order repaired and checked by reading, not compiled, not run on hardware
//
#include <hip/hip_runtime.h>
#include <math.h>

typedef __attribute__((ext_vector_type(16))) _Float16 v16h;
typedef __attribute__((ext_vector_type(8)))  _Float16 v8h;
typedef __attribute__((ext_vector_type(8)))  float    v8f;
typedef __attribute__((ext_vector_type(4)))  float    v4f;

#define LTOT 5280

__device__ __forceinline__ float cubicw(float t) {
  const float a = -0.75f;
  float at = fabsf(t);
  if (at <= 1.f) return ((a + 2.f) * at - (a + 3.f)) * at * at + 1.f;
  if (at < 2.f)  return (((a * at - 5.f * a) * at) + 8.f * a) * at - 4.f * a;
  return 0.f;
}

__device__ __forceinline__ int clampi(int v, int lo, int hi) {
  return v < lo ? lo : (v > hi ? hi : v);
}

// Fragment-order A rows: slot j (0..31) holds k = (j>>4)*8 + (j&7) + ((j>>3)&1)*16.
// Equivalently: four contiguous 8-half segments with k-bases {0,16,8,24}.
__device__ __forceinline__ int afrag_k(int j) {
  int hh = j >> 4, e = j & 15;
  return hh * 8 + (e & 7) + ((e >> 3) << 4);
}

// ---------------- head conv: 3->64, 9x9, pad 4, relu -----------------------
__global__ void conv9_in(const float* __restrict__ x, const float* __restrict__ w,
                         const float* __restrict__ bias, float* __restrict__ out) {
  int idx = blockIdx.x * blockDim.x + threadIdx.x;
  if (idx >= 4 * 64 * 1600) return;
  int p = idx % 1600; int oc = (idx / 1600) % 64; int b = idx / (1600 * 64);
  int y = p / 40, xc = p % 40;
  float s = bias[oc];
  for (int ic = 0; ic < 3; ++ic)
    for (int ky = 0; ky < 9; ++ky) {
      int yy = y + ky - 4; if ((unsigned)yy >= 40u) continue;
      for (int kx = 0; kx < 9; ++kx) {
        int xx = xc + kx - 4; if ((unsigned)xx >= 40u) continue;
        s += x[((b * 3 + ic) * 40 + yy) * 40 + xx] * w[((oc * 3 + ic) * 9 + ky) * 9 + kx];
      }
    }
  out[idx] = fmaxf(s, 0.f);
}

// ------------- generic implicit-GEMM 3x3 conv via WMMA ---------------------
// out[m, p] = bias[m] + sum_k A[m,k] * im2col(In)[k, p], K = C*9, pad 1
// block = 128 threads (4 waves); WG tile = 64M x 32N; each wave: 16M x 32N
__global__ __launch_bounds__(128)
void gemm_im2col_wmma(const float* __restrict__ A, const float* __restrict__ In,
                      const float* __restrict__ bias, float* __restrict__ Out,
                      int M, int C, int H, int W) {
  const int N = H * W;
  const int K = C * 9;
  In  += (size_t)blockIdx.z * C * N;
  Out += (size_t)blockIdx.z * M * N;
  const int n0 = blockIdx.x * 32;
  const int m0 = blockIdx.y * 64;
  const int tid = threadIdx.x;
  const int wave = tid >> 5, lane = tid & 31;
  __shared__ alignas(32) _Float16 As[64 * 32];  // [m][k], k contiguous
  __shared__ alignas(32) _Float16 Bs[32 * 32];  // [n][k] transposed
  v8f acc0 = {}, acc1 = {};
  const int Mrow = wave * 16 + (lane & 15);
  const int hh = lane >> 4;

  // ---- staging-role constants (hoisted out of the K loop) ----
  // A: thread owns row (tid>>1), segments {2*(tid&1), 2*(tid&1)+1}
  const int arow = tid >> 1;
  const int aseg0 = (tid & 1) * 2;
  const int agm = m0 + arow;
  const float* arowp = A + (size_t)(agm < M ? agm : (M - 1)) * K;
  // B: thread owns k-column (tid&31) for n = (tid>>5) + 4t, t<8
  const int bk = tid & 31;
  const int bn0 = tid >> 5;
  int by[8], bx[8];
  #pragma unroll
  for (int t = 0; t < 8; ++t) {
    int p = n0 + bn0 + 4 * t;
    by[t] = p / W; bx[t] = p - by[t] * W;
  }

  for (int kc = 0; kc < K; kc += 32) {
    // ---- A staging: 2x (two v4f loads -> one packed v8h ds_store_b128) ----
    __builtin_prefetch(arowp + kc + 32, 0, 0);
    #pragma unroll
    for (int u = 0; u < 2; ++u) {
      int seg = aseg0 + u;
      int kb = seg << 3;   // {0,8,16,24}
      v4f lo = *(const v4f*)(arowp + kc + kb);
      v4f hi = *(const v4f*)(arowp + kc + kb + 4);
      v8h hv;
      #pragma unroll
      for (int q = 0; q < 4; ++q) { hv[q] = (_Float16)lo[q]; hv[4 + q] = (_Float16)hi[q]; }
      if (agm >= M) { v8h z = {}; hv = z; }
      *(v8h*)(As + arow * 32 + seg * 8) = hv;
    }
    // ---- B staging: one (c,ki,kj) decode per chunk, 8 batched gathers ----
    {
      int gk = kc + bk;
      int c = gk / 9, r = gk - 9 * c;
      int ki = r / 3, kj = r - 3 * ki;
      const float* inc = In + (size_t)c * N;
      float vals[8];
      #pragma unroll
      for (int t = 0; t < 8; ++t) {
        int yy = by[t] + ki - 1, xx = bx[t] + kj - 1;
        bool ok = ((unsigned)yy < (unsigned)H) & ((unsigned)xx < (unsigned)W);
        int yyc = clampi(yy, 0, H - 1), xxc = clampi(xx, 0, W - 1);
        float v = inc[yyc * W + xxc];
        vals[t] = ok ? v : 0.f;
      }
      #pragma unroll
      for (int t = 0; t < 8; ++t) Bs[(bn0 + 4 * t) * 32 + bk] = (_Float16)vals[t];
    }
    __syncthreads();
    v16h af  = *(const v16h*)(As + Mrow * 32 + hh * 16);
    v16h bf0 = *(const v16h*)(Bs + (lane & 15) * 32 + hh * 16);
    v16h bf1 = *(const v16h*)(Bs + (16 + (lane & 15)) * 32 + hh * 16);
    acc0 = __builtin_amdgcn_wmma_f32_16x16x32_f16(false, af, false, bf0, (short)0, acc0, false, false);
    acc1 = __builtin_amdgcn_wmma_f32_16x16x32_f16(false, af, false, bf1, (short)0, acc1, false, false);
    __syncthreads();
  }
  int nn = lane & 15;
  int mrb = wave * 16 + (hh << 3);
  if (bias) {
    #pragma unroll
    for (int r = 0; r < 8; ++r) {
      int gm = m0 + mrb + r;
      if (gm < M) {
        float bb = bias[gm];
        Out[(size_t)gm * N + (n0 + nn)]      = acc0[r] + bb;
        Out[(size_t)gm * N + (n0 + 16 + nn)] = acc1[r] + bb;
      }
    }
  } else {
    #pragma unroll
    for (int r = 0; r < 8; ++r) {
      int gm = m0 + mrb + r;
      if (gm < M) {
        Out[(size_t)gm * N + (n0 + nn)]      = acc0[r];
        Out[(size_t)gm * N + (n0 + 16 + nn)] = acc1[r];
      }
    }
  }
}

// ---------------- depthwise 7x7 + identity (+relu / +resid) ----------------
__global__ void dwconv7_addid(const float* __restrict__ in, const float* __restrict__ w,
                              const float* __restrict__ bias, const float* __restrict__ resid,
                              float* __restrict__ out, int relu) {
  int idx = blockIdx.x * blockDim.x + threadIdx.x;
  if (idx >= 4 * 64 * 1600) return;
  int p = idx % 1600; int c = (idx / 1600) % 64; int b = idx / (1600 * 64);
  int y = p / 40, x = p % 40;
  const float* inb = in + ((size_t)b * 64 + c) * 1600;
  float s = bias[c] + inb[p];
  for (int ky = 0; ky < 7; ++ky) {
    int yy = y + ky - 3; if ((unsigned)yy >= 40u) continue;
    for (int kx = 0; kx < 7; ++kx) {
      int xx = x + kx - 3; if ((unsigned)xx >= 40u) continue;
      s += inb[yy * 40 + xx] * w[c * 49 + ky * 7 + kx];
    }
  }
  if (relu) s = fmaxf(s, 0.f);
  if (resid) s += resid[idx];
  out[idx] = s;
}

// ---------------- 1x1 conv + prelu -----------------------------------------
__global__ void conv1x1_prelu(const float* __restrict__ in, const float* __restrict__ w,
                              const float* __restrict__ bias, const float* __restrict__ alpha,
                              float* __restrict__ out, int IC, int OC, int P) {
  int idx = blockIdx.x * blockDim.x + threadIdx.x;
  if (idx >= 4 * OC * P) return;
  int p = idx % P; int oc = (idx / P) % OC; int b = idx / (P * OC);
  const float* inb = in + (size_t)b * IC * P;
  float s = bias[oc];
  for (int ic = 0; ic < IC; ++ic) s += w[oc * IC + ic] * inb[ic * P + p];
  float a = *alpha;
  out[idx] = s >= 0.f ? s : a * s;
}

// ---------------- torch-style bicubic (a=-0.75, border replicate) ----------
__global__ void bicubic_resize(const float* __restrict__ in, float* __restrict__ out,
                               int Hin, int Win, int Hout, int Wout) {
  int total = 4 * 64 * Hout * Wout;
  int idx = blockIdx.x * blockDim.x + threadIdx.x;
  if (idx >= total) return;
  int ox = idx % Wout; int oy = (idx / Wout) % Hout; int bc = idx / (Wout * Hout);
  float cy = (oy + 0.5f) * ((float)Hin / (float)Hout) - 0.5f;
  float cx = (ox + 0.5f) * ((float)Win / (float)Wout) - 0.5f;
  int iy0 = (int)floorf(cy), ix0 = (int)floorf(cx);
  const float* inb = in + (size_t)bc * Hin * Win;
  float s = 0.f;
  for (int ty = -1; ty <= 2; ++ty) {
    int iy = iy0 + ty;
    float wy = cubicw(cy - (float)iy);
    int iyc = clampi(iy, 0, Hin - 1);
    for (int tx = -1; tx <= 2; ++tx) {
      int ix = ix0 + tx;
      float wx = cubicw(cx - (float)ix);
      int ixc = clampi(ix, 0, Win - 1);
      s += wy * wx * inb[iyc * Win + ixc];
    }
  }
  out[idx] = s;
}

// ---------------- 3x3 zero-padded patch bank builder ------------------------
__global__ void extract_patches(const float* __restrict__ map, float* __restrict__ bank,
                                int C, int hs, int wss, int Loff) {
  int K = C * 9; int P = hs * wss;
  int total = 4 * P * K;
  int idx = blockIdx.x * blockDim.x + threadIdx.x;
  if (idx >= total) return;
  int k = idx % K; int p = (idx / K) % P; int b = idx / (K * P);
  int c = k / 9, r = k - 9 * c, ki = r / 3, kj = r - 3 * ki;
  int y = p / wss, x = p - y * wss;
  int yy = y + ki - 1, xx = x + kj - 1;
  bool ok = ((unsigned)yy < (unsigned)hs) & ((unsigned)xx < (unsigned)wss);
  int yyc = clampi(yy, 0, hs - 1), xxc = clampi(xx, 0, wss - 1);
  float v = map[((size_t)b * C + c) * P + yyc * wss + xxc];
  if (!ok) v = 0.f;
  bank[((size_t)b * LTOT + (Loff + p)) * K + k] = v;
}

// ---------------- per-row L2 normalize (wn = wi / max(||wi||, 1e-4)) -------
__global__ void normalize_rows(float* __restrict__ wn) {
  int idx = blockIdx.x * blockDim.x + threadIdx.x;
  if (idx >= 4 * LTOT) return;
  float* row = wn + (size_t)idx * 288;
  float ss = 0.f;
  for (int i = 0; i < 288; ++i) { float v = row[i]; ss += v * v; }
  float inv = 1.f / fmaxf(sqrtf(ss), 1e-4f);
  for (int i = 0; i < 288; ++i) row[i] *= inv;
}

// ---------------- softmax over L per spatial column (scale 10) -------------
__global__ void softmax_cols(float* __restrict__ sc) {
  __shared__ float red[256];
  int p = blockIdx.x; int tid = threadIdx.x;
  float mx = -1e30f;
  for (int l = tid; l < LTOT; l += 256) mx = fmaxf(mx, sc[(size_t)l * 1600 + p]);
  red[tid] = mx; __syncthreads();
  for (int s = 128; s > 0; s >>= 1) { if (tid < s) red[tid] = fmaxf(red[tid], red[tid + s]); __syncthreads(); }
  mx = red[0]; __syncthreads();
  float sum = 0.f;
  for (int l = tid; l < LTOT; l += 256) sum += expf(10.f * (sc[(size_t)l * 1600 + p] - mx));
  red[tid] = sum; __syncthreads();
  for (int s = 128; s > 0; s >>= 1) { if (tid < s) red[tid] += red[tid + s]; __syncthreads(); }
  float inv = 1.f / red[0];
  for (int l = tid; l < LTOT; l += 256) {
    size_t o = (size_t)l * 1600 + p;
    sc[o] = expf(10.f * (sc[o] - mx)) * inv;
  }
}

// ----- transposed-conv aggregation GEMM (M=64, N=1600, K=L*9) via WMMA -----
// hout[c,p] += 0.25 * sum_{l,u,v} ysm[l, y+1-u, x+1-v] * raw[l,c,u,v]
__global__ __launch_bounds__(128)
void attn_aggregate_wmma(const float* __restrict__ raw, const float* __restrict__ ysm,
                         float* __restrict__ hout) {
  const int n0 = blockIdx.x * 32;
  const int tid = threadIdx.x, wave = tid >> 5, lane = tid & 31;
  __shared__ alignas(32) _Float16 As[64 * 32];
  __shared__ alignas(32) _Float16 Bs[32 * 32];
  v8f acc0 = {}, acc1 = {};
  const int Mrow = wave * 16 + (lane & 15);
  const int hh = lane >> 4;
  const int K = LTOT * 9;

  // A: thread owns frag slot (tid&31) for m = (tid>>5) + 4t, t<16
  const int aj = tid & 31;
  const int am0 = tid >> 5;
  const int kperm = aj;
  // B: thread owns k-column (tid&31) for n = (tid>>5) + 4t, t<8
  const int bk = tid & 31;
  const int bn0 = tid >> 5;
  int by[8], bx[8];
  #pragma unroll
  for (int t = 0; t < 8; ++t) {
    int p = n0 + bn0 + 4 * t;
    by[t] = p / 40; bx[t] = p - by[t] * 40;
  }

  for (int kc = 0; kc < K; kc += 32) {
    // ---- A staging: one (l,r) decode, 16 batched strided loads ----
    {
      int gk = kc + kperm;
      int l = gk / 9, r = gk - 9 * l;
      const float* rp = raw + (size_t)l * 576 + r;
      float av[16];
      #pragma unroll
      for (int t = 0; t < 16; ++t) av[t] = rp[(am0 + 4 * t) * 9];
      #pragma unroll
      for (int t = 0; t < 16; ++t) As[(am0 + 4 * t) * 32 + aj] = (_Float16)av[t];
    }
    // ---- B staging: one (l,u,v) decode, 8 batched gathers ----
    {
      int gk = kc + bk;
      int l = gk / 9, r = gk - 9 * l;
      int u = r / 3, v = r - 3 * u;
      const float* yp = ysm + (size_t)l * 1600;
      float bv[8];
      #pragma unroll
      for (int t = 0; t < 8; ++t) {
        int yy = by[t] + 1 - u, xx = bx[t] + 1 - v;
        bool ok = ((unsigned)yy < 40u) & ((unsigned)xx < 40u);
        int yyc = clampi(yy, 0, 39), xxc = clampi(xx, 0, 39);
        float val = yp[yyc * 40 + xxc];
        bv[t] = ok ? val : 0.f;
      }
      #pragma unroll
      for (int t = 0; t < 8; ++t) Bs[(bn0 + 4 * t) * 32 + bk] = (_Float16)bv[t];
    }
    __syncthreads();
    v16h af  = *(const v16h*)(As + Mrow * 32 + hh * 16);
    v16h bf0 = *(const v16h*)(Bs + (lane & 15) * 32 + hh * 16);
    v16h bf1 = *(const v16h*)(Bs + (16 + (lane & 15)) * 32 + hh * 16);
    acc0 = __builtin_amdgcn_wmma_f32_16x16x32_f16(false, af, false, bf0, (short)0, acc0, false, false);
    acc1 = __builtin_amdgcn_wmma_f32_16x16x32_f16(false, af, false, bf1, (short)0, acc1, false, false);
    __syncthreads();
  }
  int nn = lane & 15;
  int mrb = wave * 16 + (hh << 3);
  #pragma unroll
  for (int r = 0; r < 8; ++r) {
    size_t o0 = (size_t)(mrb + r) * 1600 + (n0 + nn);
    size_t o1 = (size_t)(mrb + r) * 1600 + (n0 + 16 + nn);
    hout[o0] = hout[o0] + 0.25f * acc0[r];
    hout[o1] = hout[o1] + 0.25f * acc1[r];
  }
}

// ---------------- pixel shuffle r=2 + prelu ---------------------------------
__global__ void pixshuf_prelu(const float* __restrict__ in, const float* __restrict__ alpha,
                              float* __restrict__ out, int h, int w) {
  int H2 = 2 * h, W2 = 2 * w;
  int total = 4 * 64 * H2 * W2;
  int idx = blockIdx.x * blockDim.x + threadIdx.x;
  if (idx >= total) return;
  int X = idx % W2; int Y = (idx / W2) % H2; int c = (idx / (W2 * H2)) % 64; int b = idx / (W2 * H2 * 64);
  int y = Y >> 1, r1 = Y & 1, x = X >> 1, r2 = X & 1;
  float v = in[(((size_t)b * 256 + (c * 4 + r1 * 2 + r2)) * h + y) * w + x];
  float a = *alpha;
  out[idx] = v >= 0.f ? v : a * v;
}

// ---------------- tail conv: 64->3, 9x9, pad 4, (tanh+1)/2 ------------------
__global__ void conv9_out_tanh(const float* __restrict__ hin, const float* __restrict__ w,
                               const float* __restrict__ bias, float* __restrict__ out) {
  int idx = blockIdx.x * blockDim.x + threadIdx.x;
  if (idx >= 4 * 3 * 25600) return;
  int p = idx % 25600; int oc = (idx / 25600) % 3; int b = idx / (25600 * 3);
  int y = p / 160, x = p % 160;
  float s = bias[oc];
  for (int ic = 0; ic < 64; ++ic) {
    const float* ib = hin + ((size_t)b * 64 + ic) * 25600;
    const float* wb = w + ((size_t)oc * 64 + ic) * 81;
    for (int ky = 0; ky < 9; ++ky) {
      int yy = y + ky - 4; if ((unsigned)yy >= 160u) continue;
      for (int kx = 0; kx < 9; ++kx) {
        int xx = x + kx - 4; if ((unsigned)xx >= 160u) continue;
        s += ib[yy * 160 + xx] * wb[ky * 9 + kx];
      }
    }
  }
  out[idx] = 0.5f * (tanhf(s) + 1.f);
}

extern "C" void kernel_launch(void* const* d_in, const int* in_sizes, int n_in,
                              void* d_out, int out_size, void* d_ws, size_t ws_size,
                              hipStream_t stream) {
  (void)in_sizes; (void)n_in; (void)out_size; (void)ws_size;
  const float* x    = (const float*)d_in[0];
  const float* w1   = (const float*)d_in[1];
  const float* b1w  = (const float*)d_in[2];
  auto rb = [&](int i, int j) -> const float* { return (const float*)d_in[3 + (i - 2) * 6 + j]; };
  const float* b7_w  = (const float*)d_in[33]; const float* b7_b   = (const float*)d_in[34];
  const float* b7_adw= (const float*)d_in[35]; const float* b7_adb = (const float*)d_in[36];
  const float* up1_w = (const float*)d_in[37]; const float* up1_b  = (const float*)d_in[38];
  const float* up1_a = (const float*)d_in[39];
  const float* up2_w = (const float*)d_in[40]; const float* up2_b  = (const float*)d_in[41];
  const float* up2_a = (const float*)d_in[42];
  const float* out_w = (const float*)d_in[43]; const float* out_b  = (const float*)d_in[44];
  const float* mLb_w = (const float*)d_in[45]; const float* mLb_b  = (const float*)d_in[46];
  const float* mLb_a = (const float*)d_in[47];
  const float* m_w   = (const float*)d_in[48]; const float* m_b    = (const float*)d_in[49];
  const float* m_a   = (const float*)d_in[50];
  const float* asm_w = (const float*)d_in[51]; const float* asm_b  = (const float*)d_in[52];
  const float* asm_a = (const float*)d_in[53];

  float* ws  = (float*)d_ws;
  float* B1  = ws + 0;           // 409600
  float* H   = ws + 409600;      // 409600
  float* T1  = ws + 819200;      // 409600
  float* T2  = ws + 1228800;     // 409600
  float* MB  = ws + 1638400;     // 204800
  float* REF = ws + 1843200;     // 256 * 5280
  float* BASE= ws + 3194880;     // 256 * 5280
  float* MT  = ws + 4546560;     // 128 * 5280
  float* WN  = ws + 5222400;     // 4 * 5280 * 288
  float* RAW = ws + 11304960;    // 4 * 5280 * 576
  float* SC  = ws + 23470080;    // 5280 * 1600 (per-sample, reused)
  float* UP  = ws + 31918080;    // 4 * 256 * 6400
  float* H80 = ws + 38471680;    // 4 * 64 * 6400
  float* H160= SC;               // reuse: scores dead before upsampling

  conv9_in<<<dim3((409600 + 255) / 256), 256, 0, stream>>>(x, w1, b1w, B1);
  hipMemcpyAsync(H, B1, 409600 * sizeof(float), hipMemcpyDeviceToDevice, stream);

  auto resblock = [&](int i) {
    gemm_im2col_wmma<<<dim3(50, 1, 4), 128, 0, stream>>>(rb(i, 0), H, rb(i, 1), T1, 64, 64, 40, 40);
    dwconv7_addid<<<dim3((409600 + 255) / 256), 256, 0, stream>>>(T1, rb(i, 4), rb(i, 5), nullptr, T2, 1);
    gemm_im2col_wmma<<<dim3(50, 1, 4), 128, 0, stream>>>(rb(i, 2), T2, rb(i, 3), T1, 64, 64, 40, 40);
    dwconv7_addid<<<dim3((409600 + 255) / 256), 256, 0, stream>>>(T1, rb(i, 4), rb(i, 5), H, H, 0);
  };
  resblock(2); resblock(3); resblock(4);

  // ---- pyramid attention ----
  conv1x1_prelu<<<dim3((4 * 32 * 1600 + 255) / 256), 256, 0, stream>>>(H, mLb_w, mLb_b, mLb_a, MB, 64, 32, 1600);
  const int hs[5]   = {40, 36, 32, 28, 24};
  const int loff[5] = {0, 1600, 2896, 3920, 4704};
  for (int s = 0; s < 5; ++s) {
    int P = hs[s] * hs[s];
    float* ref = REF  + (size_t)256 * loff[s];
    float* bas = BASE + (size_t)256 * loff[s];
    float* mts = MT   + (size_t)128 * loff[s];
    bicubic_resize<<<dim3((4 * 64 * P + 255) / 256), 256, 0, stream>>>(H, ref, 40, 40, hs[s], hs[s]);
    conv1x1_prelu<<<dim3((4 * 64 * P + 255) / 256), 256, 0, stream>>>(ref, asm_w, asm_b, asm_a, bas, 64, 64, P);
    conv1x1_prelu<<<dim3((4 * 32 * P + 255) / 256), 256, 0, stream>>>(ref, m_w, m_b, m_a, mts, 64, 32, P);
    extract_patches<<<dim3((4 * P * 576 + 255) / 256), 256, 0, stream>>>(bas, RAW, 64, hs[s], hs[s], loff[s]);
    extract_patches<<<dim3((4 * P * 288 + 255) / 256), 256, 0, stream>>>(mts, WN, 32, hs[s], hs[s], loff[s]);
  }
  normalize_rows<<<dim3((4 * LTOT + 255) / 256), 256, 0, stream>>>(WN);
  for (int b = 0; b < 4; ++b) {
    gemm_im2col_wmma<<<dim3(50, 83, 1), 128, 0, stream>>>(WN + (size_t)b * LTOT * 288,
                                                          MB + (size_t)b * 32 * 1600,
                                                          nullptr, SC, LTOT, 32, 40, 40);
    softmax_cols<<<dim3(1600), 256, 0, stream>>>(SC);
    attn_aggregate_wmma<<<dim3(50), 128, 0, stream>>>(RAW + (size_t)b * LTOT * 576, SC,
                                                      H + (size_t)b * 64 * 1600);
  }

  resblock(5); resblock(6);

  gemm_im2col_wmma<<<dim3(50, 1, 4), 128, 0, stream>>>(b7_w, H, b7_b, T1, 64, 64, 40, 40);
  dwconv7_addid<<<dim3((409600 + 255) / 256), 256, 0, stream>>>(T1, b7_adw, b7_adb, B1, H, 0);

  gemm_im2col_wmma<<<dim3(50, 4, 4), 128, 0, stream>>>(up1_w, H, up1_b, UP, 256, 64, 40, 40);
  pixshuf_prelu<<<dim3((4 * 64 * 6400 + 255) / 256), 256, 0, stream>>>(UP, up1_a, H80, 40, 40);
  gemm_im2col_wmma<<<dim3(200, 4, 4), 128, 0, stream>>>(up2_w, H80, up2_b, UP, 256, 64, 80, 80);
  pixshuf_prelu<<<dim3((4 * 64 * 25600 + 255) / 256), 256, 0, stream>>>(UP, up2_a, H160, 80, 80);

  conv9_out_tanh<<<dim3((4 * 3 * 25600 + 255) / 256), 256, 0, stream>>>(H160, out_w, out_b, (float*)d_out);
}
